// EEGHybridNet_80470507258219
// MI455X (gfx1250) — matrix operand order repaired and checked by reading, not compiled, not run on hardware
//
#include <hip/hip_runtime.h>
#include <math.h>

typedef __attribute__((ext_vector_type(16))) _Float16 v16h;
typedef __attribute__((ext_vector_type(8)))  _Float16 v8h;
typedef __attribute__((ext_vector_type(8)))  float    v8f;

#define WMMA_F16(A, B, C) \
    __builtin_amdgcn_wmma_f32_16x16x32_f16(false, (A), false, (B), (short)0, (C), false, false)

// ---------------------------------------------------------------- workspace layout
static constexpr size_t OFF_COMB = 0;                          // f16 [256][1024][78]
static constexpr size_t SZ_COMB  = (size_t)256 * 1024 * 78 * 2;
static constexpr size_t OFF_Y    = OFF_COMB + SZ_COMB;         // f16 [256][512][64]
static constexpr size_t SZ_Y     = (size_t)256 * 512 * 64 * 2;
static constexpr size_t OFF_WC   = OFF_Y + SZ_Y;               // f16 [64][384]  K' = r*128+ic
static constexpr size_t OFF_WIH  = OFF_WC  + (size_t)64 * 384 * 2;   // f16 [64][128] (W_ih^T)
static constexpr size_t OFF_WHH  = OFF_WIH + (size_t)64 * 128 * 2;   // f16 [32][128] (W_hh^T)
static constexpr size_t OFF_BIAS = OFF_WHH + (size_t)32 * 128 * 2;   // f32 [128]

// ---------------------------------------------------------------- helpers
__device__ __forceinline__ float sigf(float x) { return 1.0f / (1.0f + __expf(-x)); }

// A-fragment (16x32 f16): lane<16 -> M=lane, K = 16h + e (contiguous); h = lane>>4.
__device__ __forceinline__ v16h load_frag_row(const _Float16* __restrict__ row, int half) {
    v16h a;
    ((v8h*)&a)[0] = *(const v8h*)(row + 16 * half);
    ((v8h*)&a)[1] = *(const v8h*)(row + 16 * half + 8);
    return a;
}

// B-fragment (32x16 f16) from 16 CONTIGUOUS halfs: element e -> K = 16*half + e.
__device__ __forceinline__ v16h load_frag_b16(const _Float16* __restrict__ p) {
    v16h b;
    ((v8h*)&b)[0] = *(const v8h*)(p);
    ((v8h*)&b)[1] = *(const v8h*)(p + 8);
    return b;
}

// db4 single-level DWT, pywt 'symmetric' mode, fully unrolled (compile-time N).
template <int N>
__device__ __forceinline__ void dwt_level(const float* a, float* cA, float* cD) {
    constexpr float LO[8] = {-0.010597401784997278f, 0.032883011666982945f,
                             0.030841381835986965f, -0.18703481171888114f,
                             -0.02798376941698385f,  0.6308807679295904f,
                              0.7148465705525415f,   0.23037781330885523f};
    constexpr float HI[8] = {-0.23037781330885523f,  0.7148465705525415f,
                             -0.6308807679295904f,  -0.02798376941698385f,
                              0.18703481171888114f,  0.030841381835986965f,
                             -0.032883011666982945f,-0.010597401784997278f};
    constexpr int L = (N + 7) / 2;
#pragma unroll
    for (int k = 0; k < L; ++k) {
        float sa = 0.f, sd = 0.f;
#pragma unroll
        for (int j = 0; j < 8; ++j) {
            int idx = 2 * k + 1 - j;                       // matches pywt phase
            int m = (idx < 0) ? (-idx - 1) : ((idx >= N) ? (2 * N - 1 - idx) : idx);
            float v = a[m];
            sa += LO[j] * v;
            sd += HI[j] * v;
        }
        cA[k] = sa; cD[k] = sd;
    }
}

// ---------------------------------------------------------------- kernel 0: weight prep
__global__ void prep_kernel(const float* __restrict__ conv_w,   // [64][78][3]
                            const float* __restrict__ w_ih,     // [128][64]
                            const float* __restrict__ w_hh,     // [128][32]
                            const float* __restrict__ b_ih, const float* __restrict__ b_hh,
                            _Float16* __restrict__ Wc, _Float16* __restrict__ Wih,
                            _Float16* __restrict__ Whh, float* __restrict__ bias) {
    int tid = threadIdx.x;
    for (int e = tid; e < 64 * 384; e += 256) {     // K' = r*128 + ic (ic 78..127 zero)
        int oc = e / 384, kk = e - oc * 384;
        int r = kk >> 7, ic = kk & 127;
        Wc[e] = (ic < 78) ? (_Float16)conv_w[(oc * 78 + ic) * 3 + r] : (_Float16)0.0f;
    }
    for (int e = tid; e < 64 * 128; e += 256) {     // W_ih^T [K=64][N=128]
        int k = e >> 7, n = e & 127;
        Wih[e] = (_Float16)w_ih[n * 64 + k];
    }
    for (int e = tid; e < 32 * 128; e += 256) {     // W_hh^T [K=32][N=128]
        int k = e >> 7, n = e & 127;
        Whh[e] = (_Float16)w_hh[n * 32 + k];
    }
    if (tid < 128) bias[tid] = b_ih[tid] + b_hh[tid];
}

// ---------------------------------------------------------------- kernel 1: features
// One thread per (b,s): raw(21) + |rfft|(11) + wavedec4(46) = 78 channels, contiguous.
__global__ void feat_kernel(const float* __restrict__ x, _Float16* __restrict__ comb) {
    int gid = blockIdx.x * 256 + threadIdx.x;
    int b = gid >> 10, s = gid & 1023;
    const float* xp = x + ((size_t)b * 1024 + s) * 21;
    float v[21];
#pragma unroll
    for (int t = 0; t < 21; ++t) v[t] = xp[t];

    float fo[78];                                   // feature vector, then one burst store
#pragma unroll
    for (int t = 0; t < 21; ++t) fo[t] = v[t];

    // DFT magnitudes: only 21 distinct twiddles (k*t mod 21) -> 21 sincos total
    float ct[21], st[21];
#pragma unroll
    for (int m = 0; m < 21; ++m) {
        float sn, cs;
        __sincosf(-0.2991993003418851f * (float)m, &sn, &cs);   // -2*pi/21 * m
        ct[m] = cs; st[m] = sn;
    }
#pragma unroll
    for (int k = 0; k < 11; ++k) {
        float re = 0.f, im = 0.f;
#pragma unroll
        for (int t = 0; t < 21; ++t) {
            constexpr_int:;
            const int idx = (k * t) % 21;           // compile-time after unroll
            re += v[t] * ct[idx];
            im += v[t] * st[idx];
        }
        fo[21 + k] = sqrtf(re * re + im * im);
    }

    // wavedec level 4: 21 -> 14 -> 10 -> 8 -> 7; concat [cA4,cD4,cD3,cD2,cD1]
    float a1[14], d1[14], a2[10], d2[10], a3[8], d3[8], a4[7], d4[7];
    dwt_level<21>(v, a1, d1);
    dwt_level<14>(a1, a2, d2);
    dwt_level<10>(a2, a3, d3);
    dwt_level<8>(a3, a4, d4);
#pragma unroll
    for (int k = 0; k < 7; ++k)  fo[32 + k] = a4[k];
#pragma unroll
    for (int k = 0; k < 7; ++k)  fo[39 + k] = d4[k];
#pragma unroll
    for (int k = 0; k < 8; ++k)  fo[46 + k] = d3[k];
#pragma unroll
    for (int k = 0; k < 10; ++k) fo[54 + k] = d2[k];
#pragma unroll
    for (int k = 0; k < 14; ++k) fo[64 + k] = d1[k];

    _Float16* cb = comb + ((size_t)b * 1024 + s) * 78;          // channel-contiguous
#pragma unroll
    for (int c = 0; c < 78; ++c) cb[c] = (_Float16)fo[c];
}

// ---------------------------------------------------------------- kernel 2: conv+relu+pool
// Block = (batch, 64-wide s tile). GEMM: M=64 oc, N=64 s, K'=384 (= r*128+ic, zero-padded).
// LDS tile is TRANSPOSED [s][ch] so each B fragment is two contiguous 16B LDS loads:
// B elem e -> K = kc*32 + 16*half + e -> r = (kc*32)>>7 (chunk-uniform, compile time),
// ch = ((kc*32)&127) + 16*half + e (contiguous in e). No division, no divergence.
__global__ void conv_kernel(const _Float16* __restrict__ comb, const _Float16* __restrict__ Wc,
                            const float* __restrict__ conv_b, _Float16* __restrict__ y) {
    __shared__ __align__(16) _Float16 sCombT[66 * 128];  // [s_local+halo][ch], ch 78..127 = 0
    __shared__ float sOut[64 * 65];                      // [oc][s_local], stride 65

    const int tid  = threadIdx.x;
    const int b    = blockIdx.x >> 4;
    const int s0   = (blockIdx.x & 15) << 6;
    const int wave = tid >> 5, lane = tid & 31;
    const int half = lane >> 4, lm = lane & 15;

    // stage transposed halo tile (global s = s0-1 .. s0+64), zero-padded everywhere else
    for (int e = tid; e < 66 * 128; e += 256) {
        int si = e >> 7, ch = e & 127;
        int gs = s0 - 1 + si;
        _Float16 val = (_Float16)0.0f;
        if (ch < 78 && gs >= 0 && gs < 1024)
            val = comb[((size_t)b * 1024 + gs) * 78 + ch];
        sCombT[e] = val;
    }
    __syncthreads();

    const int mt  = wave >> 1;                 // oc tile
    const int ntb = (wave & 1) * 2;            // first of two s tiles
    const int sl0 = ntb * 16 + lm;             // B column (s_local) for frag 0
    const int sl1 = sl0 + 16;
    v8f acc0 = {}, acc1 = {};

#pragma unroll
    for (int kc = 0; kc < 12; ++kc) {          // K' = 12 x 32
        const int r   = (kc * 32) >> 7;        // compile-time per chunk
        const int icc = (kc * 32) & 127;       // compile-time per chunk
        v16h a  = load_frag_row(Wc + (size_t)(mt * 16 + lm) * 384 + kc * 32, half);
        v16h b0 = load_frag_b16(&sCombT[(sl0 + r) * 128 + icc + 16 * half]);
        v16h b1 = load_frag_b16(&sCombT[(sl1 + r) * 128 + icc + 16 * half]);
        acc0 = WMMA_F16(a, b0, acc0);
        acc1 = WMMA_F16(a, b1, acc1);
    }

    // C/D layout: row M = v + 8*half (+ mt*16), col N = lm (+ nt*16). Fuse bias.
#pragma unroll
    for (int v = 0; v < 8; ++v) {
        int oc = mt * 16 + v + 8 * half;
        float bv = conv_b[oc];
        sOut[oc * 65 + sl0] = acc0[v] + bv;
        sOut[oc * 65 + sl1] = acc1[v] + bv;
    }
    __syncthreads();

    // ReLU + maxpool(2); y layout [B][512][64], oc contiguous for LSTM A-frags
    for (int e = tid; e < 64 * 32; e += 256) {
        int oc = e >> 5, ps = e & 31;
        float m = fmaxf(fmaxf(sOut[oc * 65 + 2 * ps], sOut[oc * 65 + 2 * ps + 1]), 0.0f);
        y[((size_t)b * 512 + (s0 >> 1) + ps) * 64 + oc] = (_Float16)m;
    }
}

// ---------------------------------------------------------------- kernel 3: LSTM + FC
// 16 blocks x 16 batch rows; 8 waves each own one 16-col gate tile (gates = 128 cols).
__global__ void lstm_kernel(const _Float16* __restrict__ y, const _Float16* __restrict__ Wih,
                            const _Float16* __restrict__ Whh, const float* __restrict__ bias,
                            const float* __restrict__ fc_w, const float* __restrict__ fc_b,
                            float* __restrict__ out) {
    __shared__ float sGates[16 * 132];                 // [batch16][gates128], stride 132
    __shared__ __align__(16) _Float16 sH[16 * 32];     // h (f16) for next-step WMMA A
    __shared__ float sHf[16 * 32];                     // h (f32) for final FC

    const int tid  = threadIdx.x;
    const int b0   = blockIdx.x * 16;
    const int wave = tid >> 5, lane = tid & 31;
    const int half = lane >> 4, lm = lane & 15;
    const int n0   = wave * 16;                        // gate column tile

    sH[tid]        = (_Float16)0.0f; sH[tid + 256]  = (_Float16)0.0f;
    sHf[tid]       = 0.0f;           sHf[tid + 256] = 0.0f;
    float c0 = 0.0f, c1 = 0.0f;                        // cell state: 2 entries/thread

    // loop-invariant B fragments (element e -> K = 16*half + e)
    v16h bx0, bx1, bh;
#pragma unroll
    for (int e = 0; e < 16; ++e) {
        int k = 16 * half + e;
        bx0[e] = Wih[(size_t)k * 128 + n0 + lm];            // K 0..31 of W_ih^T
        bx1[e] = Wih[(size_t)(32 + k) * 128 + n0 + lm];     // K 32..63
        bh[e]  = Whh[(size_t)k * 128 + n0 + lm];            // K 0..31 of W_hh^T
    }
    const float bias_n = bias[n0 + lm];
    __syncthreads();

    for (int t = 0; t < 512; ++t) {
        const _Float16* yp = y + ((size_t)(b0 + lm) * 512 + t) * 64;
        __builtin_prefetch(yp + 64, 0, 0);                  // next timestep
        v16h ax0 = load_frag_row(yp, half);
        v16h ax1 = load_frag_row(yp + 32, half);
        v16h ah  = load_frag_row(&sH[lm * 32], half);       // A_h from LDS h-buffer

        v8f acc = {};
        acc = WMMA_F16(ax0, bx0, acc);
        acc = WMMA_F16(ax1, bx1, acc);
        acc = WMMA_F16(ah,  bh,  acc);

#pragma unroll
        for (int v = 0; v < 8; ++v)
            sGates[(v + 8 * half) * 132 + n0 + lm] = acc[v] + bias_n;
        __syncthreads();

#pragma unroll
        for (int q = 0; q < 2; ++q) {
            int idx = tid + q * 256;
            int bl = idx >> 5, hd = idx & 31;
            const float* g = &sGates[bl * 132];
            float gi = g[hd], gf = g[32 + hd], gg = g[64 + hd], go = g[96 + hd];
            float& c = q ? c1 : c0;
            c = sigf(gf) * c + sigf(gi) * tanhf(gg);
            float h = sigf(go) * tanhf(c);
            sH[bl * 32 + hd]  = (_Float16)h;
            sHf[bl * 32 + hd] = h;
        }
        __syncthreads();
    }

    if (tid < 16) {                                         // FC: [16,32] @ [32,1] + b
        float acc = fc_b[0];
#pragma unroll
        for (int h = 0; h < 32; ++h) acc += sHf[tid * 32 + h] * fc_w[h];
        out[b0 + tid] = acc;
    }
}

// ---------------------------------------------------------------- launcher
extern "C" void kernel_launch(void* const* d_in, const int* in_sizes, int n_in,
                              void* d_out, int out_size, void* d_ws, size_t ws_size,
                              hipStream_t stream) {
    (void)in_sizes; (void)n_in; (void)out_size; (void)ws_size;
    const float* x      = (const float*)d_in[0];
    const float* conv_w = (const float*)d_in[1];
    const float* conv_b = (const float*)d_in[2];
    const float* w_ih   = (const float*)d_in[3];
    const float* w_hh   = (const float*)d_in[4];
    const float* b_ih   = (const float*)d_in[5];
    const float* b_hh   = (const float*)d_in[6];
    const float* fc_w   = (const float*)d_in[7];
    const float* fc_b   = (const float*)d_in[8];
    float* out = (float*)d_out;

    char* ws = (char*)d_ws;
    _Float16* comb = (_Float16*)(ws + OFF_COMB);
    _Float16* y    = (_Float16*)(ws + OFF_Y);
    _Float16* Wc   = (_Float16*)(ws + OFF_WC);
    _Float16* Wih  = (_Float16*)(ws + OFF_WIH);
    _Float16* Whh  = (_Float16*)(ws + OFF_WHH);
    float*    bias = (float*)(ws + OFF_BIAS);

    prep_kernel<<<1, 256, 0, stream>>>(conv_w, w_ih, w_hh, b_ih, b_hh, Wc, Wih, Whh, bias);
    feat_kernel<<<1024, 256, 0, stream>>>(x, comb);
    conv_kernel<<<4096, 256, 0, stream>>>(comb, Wc, conv_b, y);
    lstm_kernel<<<16, 256, 0, stream>>>(y, Wih, Whh, bias, fc_w, fc_b, out);
}
